// MultiHeadSelfAttention_61529701482709
// MI455X (gfx1250) — matrix operand order repaired and checked by reading, not compiled, not run on hardware
//
#include <hip/hip_runtime.h>
#include <hip/hip_bf16.h>
#include <math.h>

// Problem constants (match reference)
#define BATCH  2
#define SEQ    2048
#define DMODEL 1024
#define NHEAD  16
#define DHEAD  64

typedef __bf16 v16bf __attribute__((ext_vector_type(16)));
typedef __bf16 v8bf  __attribute__((ext_vector_type(8)));
typedef float  v8f   __attribute__((ext_vector_type(8)));
typedef int    v4i   __attribute__((ext_vector_type(4)));

static __device__ __forceinline__ v16bf cat8(v8bf lo, v8bf hi) {
  return __builtin_shufflevector(lo, hi, 0,1,2,3,4,5,6,7,8,9,10,11,12,13,14,15);
}

static __device__ __forceinline__ v8f wmma_bf16(v16bf a, v16bf b, v8f c) {
  // D = A(16x32 bf16) * B(32x16 bf16) + C(16x16 f32)
  return __builtin_amdgcn_wmma_f32_16x16x32_bf16(false, a, false, b, (short)0, c,
                                                 false, false);
}

// ---------------------------------------------------------------------------
// Async global->LDS copy (CDNA5 GLOBAL_LOAD_ASYNC_TO_LDS_B128, ASYNCcnt).
// Builtin parameters are (v4i AS1*, v4i AS3*, imm offset, imm cpol) per the
// round-2 diagnostic. Falls back to a synchronous 16B copy if absent.
// ---------------------------------------------------------------------------
#if __has_builtin(__builtin_amdgcn_global_load_async_to_lds_b128)
#define HAVE_ASYNC_LDS 1
typedef __attribute__((address_space(1))) v4i* gv4i_ptr;
typedef __attribute__((address_space(3))) v4i* lv4i_ptr;
#else
#define HAVE_ASYNC_LDS 0
#endif

static __device__ __forceinline__ void copy16_g2l(const __bf16* g, __bf16* l) {
#if HAVE_ASYNC_LDS
  __builtin_amdgcn_global_load_async_to_lds_b128((gv4i_ptr)g, (lv4i_ptr)l, 0, 0);
#else
  *(v8bf*)l = *(const v8bf*)g;
#endif
}

static __device__ __forceinline__ void wait_async_done() {
#if HAVE_ASYNC_LDS
  asm volatile("s_wait_asynccnt 0" ::: "memory");
#endif
}

// ---------------------------------------------------------------------------
// fp32 -> bf16 conversion
// ---------------------------------------------------------------------------
__global__ void cvt_f32_to_bf16(const float* __restrict__ in,
                                __bf16* __restrict__ out, int n) {
  int i = blockIdx.x * blockDim.x + threadIdx.x;
  if (i < n) out[i] = (__bf16)in[i];
}

// ---------------------------------------------------------------------------
// C[M,N] = A[M,K](bf16) @ W[N,K]^T(bf16); fp32 or bf16 output.
// Block: 256 threads = 8 waves; block tile 128x128; wave tile 32x64
// (waves 4 in M x 2 in N; 8 WMMAs per wave per 32-k step).
// A/B panels (128x32) double-buffered in LDS, filled with async-to-LDS loads,
// padded row stride 40 bf16 (80B) => 16B-aligned, bank-conflict-free reads.
// Pipeline per step: wait(asynccnt) -> barrier -> issue next panel -> compute.
// ---------------------------------------------------------------------------
static constexpr int LDSP = 40;   // padded LDS row stride in bf16 elements

template <bool F32OUT>
__global__ void __launch_bounds__(256)
gemm_bf16_wmma(const __bf16* __restrict__ A,
               const __bf16* __restrict__ W,
               float* __restrict__ Cf,
               __bf16* __restrict__ Cb,
               int M, int N, int K) {
  __shared__ __bf16 sA[2][128 * LDSP];   // 20 KB
  __shared__ __bf16 sB[2][128 * LDSP];   // 20 KB

  const int tid  = threadIdx.x;
  const int lane = tid & 31;
  const int wave = tid >> 5;
  const int half = lane >> 4;   // 0: lanes 0-15, 1: lanes 16-31
  const int l16  = lane & 15;
  const int wm = wave & 3;      // 4 waves along M
  const int wn = wave >> 2;     // 2 waves along N
  const int blockM = blockIdx.x * 128;
  const int blockN = blockIdx.y * 128;

  // Panel fill: 128 rows x 32 k (8 KB each); 256 threads x 2 passes x 16 B.
  auto stage = [&](int buf, int k0) {
#pragma unroll
    for (int p = 0; p < 2; ++p) {
      int linear = p * 256 + tid;
      int row = linear >> 2;          // 0..127
      int ch  = (linear & 3) * 8;     // 0,8,16,24
      copy16_g2l(A + (size_t)(blockM + row) * K + k0 + ch,
                 &sA[buf][row * LDSP + ch]);
      copy16_g2l(W + (size_t)(blockN + row) * K + k0 + ch,
                 &sB[buf][row * LDSP + ch]);
    }
  };

  v8f acc[2][4] = {};
  const int NK = K / 32;

  stage(0, 0);
  for (int ks = 0; ks < NK; ++ks) {
    wait_async_done();              // my panel-ks data has landed in LDS
    __syncthreads();                // everyone's data landed; prev compute done
    if (ks + 1 < NK) stage((ks + 1) & 1, (ks + 1) * 32);

    const __bf16* a0 = &sA[ks & 1][0];
    const __bf16* b0 = &sB[ks & 1][0];

    // A fragments (2 per wave): row = wm*32 + mt*16 + l16, K chunks {h*8, h*8+16}
    v16bf af[2];
#pragma unroll
    for (int mt = 0; mt < 2; ++mt) {
      const __bf16* ap = a0 + (wm * 32 + mt * 16 + l16) * LDSP + half * 8;
      af[mt] = cat8(*(const v8bf*)ap, *(const v8bf*)(ap + 16));
    }
    // B fragments (4 per wave): col = wn*64 + f*16 + l16, K chunks {h*8, h*8+16}
#pragma unroll
    for (int f = 0; f < 4; ++f) {
      const __bf16* bp = b0 + (wn * 64 + f * 16 + l16) * LDSP + half * 8;
      v16bf bfrag = cat8(*(const v8bf*)bp, *(const v8bf*)(bp + 16));
#pragma unroll
      for (int mt = 0; mt < 2; ++mt)
        acc[mt][f] = wmma_bf16(af[mt], bfrag, acc[mt][f]);
    }
  }

#pragma unroll
  for (int mt = 0; mt < 2; ++mt) {
#pragma unroll
    for (int f = 0; f < 4; ++f) {
#pragma unroll
      for (int r = 0; r < 8; ++r) {
        size_t row = (size_t)(blockM + wm * 32 + mt * 16 + half * 8 + r);
        size_t col = (size_t)(blockN + wn * 64 + f * 16 + l16);
        if (F32OUT) Cf[row * N + col] = acc[mt][f][r];
        else        Cb[row * N + col] = (__bf16)acc[mt][f][r];
      }
    }
  }
}

// ---------------------------------------------------------------------------
// RoPE on Q and K + rearrange [B,S,H,DH] -> [B,H,S,DH].
// ---------------------------------------------------------------------------
__global__ void rope_rearrange_qk(const __bf16* __restrict__ qraw,
                                  const __bf16* __restrict__ kraw,
                                  const int* __restrict__ pos,
                                  __bf16* __restrict__ qout,
                                  __bf16* __restrict__ kout) {
  int idx = blockIdx.x * blockDim.x + threadIdx.x;
  if (idx >= BATCH * NHEAD * SEQ * (DHEAD / 2)) return;
  int i  = idx & 31;
  int s  = (idx >> 5) & (SEQ - 1);
  int bh = idx >> 16;
  int b  = bh >> 4;
  int h  = bh & 15;
  size_t inoff  = ((size_t)b * SEQ + s) * DMODEL + (size_t)h * DHEAD + 2 * i;
  size_t outoff = ((size_t)bh * SEQ + s) * DHEAD + 2 * i;

  float p    = (float)pos[s];
  float freq = __expf(-(float)i * (9.2103403720f / 32.0f)); // 10000^(-i/32)
  float ang  = p * freq;
  float c = __cosf(ang), sn = __sinf(ang);
  {
    float x1 = (float)qraw[inoff], x2 = (float)qraw[inoff + 1];
    qout[outoff]     = (__bf16)(x1 * c - x2 * sn);
    qout[outoff + 1] = (__bf16)(x1 * sn + x2 * c);
  }
  {
    float x1 = (float)kraw[inoff], x2 = (float)kraw[inoff + 1];
    kout[outoff]     = (__bf16)(x1 * c - x2 * sn);
    kout[outoff + 1] = (__bf16)(x1 * sn + x2 * c);
  }
}

// ---------------------------------------------------------------------------
// V rearrange [B,S,H,DH] -> [B,H,DH,S]
// ---------------------------------------------------------------------------
__global__ void transpose_v(const __bf16* __restrict__ vraw,
                            __bf16* __restrict__ vt) {
  int idx = blockIdx.x * blockDim.x + threadIdx.x;
  if (idx >= BATCH * NHEAD * SEQ * DHEAD) return;
  int d  = idx & 63;
  int s  = (idx >> 6) & (SEQ - 1);
  int bh = idx >> 17;
  int b  = bh >> 4;
  int h  = bh & 15;
  vt[((size_t)bh * DHEAD + d) * SEQ + s] =
      vraw[((size_t)b * SEQ + s) * DMODEL + (size_t)h * DHEAD + d];
}

// ---------------------------------------------------------------------------
// Causal flash attention. One wave per 32-query block (2 q-tiles), 4 waves
// per 128-thread block. K / V fragments are loaded once per 32-key chunk and
// reused by both q-tiles (halves K/V traffic vs 16-query waves).
// Per chunk: 8 score WMMAs, online softmax, P via per-wave LDS buffers
// (C-layout -> A-layout), 8 PV WMMAs.
// ---------------------------------------------------------------------------
__global__ void __launch_bounds__(128)
attn_flash_wmma(const __bf16* __restrict__ qb,
                const __bf16* __restrict__ kb,
                const __bf16* __restrict__ vt,
                __bf16* __restrict__ ctx) {
  const int lane = threadIdx.x & 31;
  const int wave = threadIdx.x >> 5;
  const int half = lane >> 4;
  const int l16  = lane & 15;
  const int wid  = blockIdx.x * 4 + wave;
  const int qblk = wid & (SEQ / 32 - 1);   // 64 query blocks per (b,h)
  const int bh   = wid >> 6;
  const int b    = bh >> 4;
  const int h    = bh & 15;
  const int q0   = qblk * 32;

  const __bf16* Q  = qb + (size_t)bh * SEQ * DHEAD;
  const __bf16* Kp = kb + (size_t)bh * SEQ * DHEAD;
  const __bf16* Vp = vt + (size_t)bh * DHEAD * SEQ;

  __shared__ __bf16 pls[4][2][16 * 32];   // per-wave P staging (8 KB)

  // Q fragments: 2 q-tiles x 2 k-steps
  v16bf qa[2][2];
#pragma unroll
  for (int qt = 0; qt < 2; ++qt) {
#pragma unroll
    for (int st = 0; st < 2; ++st) {
      const __bf16* p = Q + (size_t)(q0 + qt * 16 + l16) * DHEAD + st * 32 + half * 16;
      qa[qt][st] = cat8(*(const v8bf*)p, *(const v8bf*)(p + 8));
    }
  }

  v8f acc[2][4] = {};
  float mrun[2][8], lrun[2][8];
#pragma unroll
  for (int qt = 0; qt < 2; ++qt)
#pragma unroll
    for (int r = 0; r < 8; ++r) { mrun[qt][r] = -1e30f; lrun[qt][r] = 0.0f; }

  const float scale = 0.125f;   // 1/sqrt(DH)
  const int kend = q0 + 31;

  for (int j0 = 0; j0 <= kend; j0 += 32) {
    // scores: 2 q-tiles x 2 key-tiles; K fragments shared across q-tiles
    v8f s[2][2] = {};
#pragma unroll
    for (int st = 0; st < 2; ++st) {
      v16bf k0f = *(const v16bf*)(Kp + (size_t)(j0 + l16) * DHEAD + st * 32 + half * 16);
      v16bf k1f = *(const v16bf*)(Kp + (size_t)(j0 + 16 + l16) * DHEAD + st * 32 + half * 16);
#pragma unroll
      for (int qt = 0; qt < 2; ++qt) {
        s[qt][0] = wmma_bf16(qa[qt][st], k0f, s[qt][0]);
        s[qt][1] = wmma_bf16(qa[qt][st], k1f, s[qt][1]);
      }
    }

    // online softmax per row; row m = half*8 + r within each q-tile
#pragma unroll
    for (int qt = 0; qt < 2; ++qt) {
#pragma unroll
      for (int r = 0; r < 8; ++r) {
        int qrow = q0 + qt * 16 + half * 8 + r;
        float v0 = s[qt][0][r] * scale;
        float v1 = s[qt][1][r] * scale;
        if (j0 + l16 > qrow)      v0 = -1e30f;   // causal mask
        if (j0 + 16 + l16 > qrow) v1 = -1e30f;
        float mx = fmaxf(v0, v1);
        mx = fmaxf(mx, __shfl_xor(mx, 1));
        mx = fmaxf(mx, __shfl_xor(mx, 2));
        mx = fmaxf(mx, __shfl_xor(mx, 4));
        mx = fmaxf(mx, __shfl_xor(mx, 8));
        float mnew = fmaxf(mrun[qt][r], mx);
        float corr = __expf(mrun[qt][r] - mnew);
        mrun[qt][r] = mnew;
        float p0 = __expf(v0 - mnew);
        float p1 = __expf(v1 - mnew);
        float ps = p0 + p1;
        ps += __shfl_xor(ps, 1);
        ps += __shfl_xor(ps, 2);
        ps += __shfl_xor(ps, 4);
        ps += __shfl_xor(ps, 8);
        lrun[qt][r] = lrun[qt][r] * corr + ps;
#pragma unroll
        for (int f = 0; f < 4; ++f) acc[qt][f][r] *= corr;
        int m = half * 8 + r;
        pls[wave][qt][m * 32 + l16]      = (__bf16)p0;
        pls[wave][qt][m * 32 + 16 + l16] = (__bf16)p1;
      }
    }

    asm volatile("s_wait_dscnt 0" ::: "memory");  // same-wave store->load

    // reload P tiles as 16x32 A-fragments
    v16bf pa[2];
#pragma unroll
    for (int qt = 0; qt < 2; ++qt)
      pa[qt] = cat8(*(const v8bf*)&pls[wave][qt][l16 * 32 + half * 16],
                    *(const v8bf*)&pls[wave][qt][l16 * 32 + half * 16 + 8]);

    // ctx += P @ V; V fragments shared across q-tiles
#pragma unroll
    for (int f = 0; f < 4; ++f) {
      v16bf vb = *(const v16bf*)(Vp + (size_t)(f * 16 + l16) * SEQ + j0 + half * 16);
#pragma unroll
      for (int qt = 0; qt < 2; ++qt)
        acc[qt][f] = wmma_bf16(pa[qt], vb, acc[qt][f]);
    }
  }

  // normalize and scatter to [B,S,D] bf16
#pragma unroll
  for (int qt = 0; qt < 2; ++qt) {
#pragma unroll
    for (int f = 0; f < 4; ++f) {
#pragma unroll
      for (int r = 0; r < 8; ++r) {
        size_t row = (size_t)b * SEQ + q0 + qt * 16 + half * 8 + r;
        size_t col = (size_t)h * DHEAD + f * 16 + l16;
        ctx[row * DMODEL + col] = (__bf16)(acc[qt][f][r] / lrun[qt][r]);
      }
    }
  }
}

// ---------------------------------------------------------------------------
// Host-side orchestration. Workspace layout (requires ws_size >= 72 MiB):
//   [ 0, 8M)   xb              bf16 x         (B*S*D)
//   [ 8M,16M)  wqb/wkb/wvb/wob bf16 weights   (4 * D*D)
//   [16M,24M)  qraw  [24M,32M) kraw  [32M,40M) vraw   (bf16, [B,S,D])
//   [40M,48M)  qrope [48M,56M) krope                  (bf16, [B,H,S,DH])
//   [56M,64M)  vt                                     (bf16, [B,H,DH,S])
//   [64M,72M)  ctx                                    (bf16, [B,S,D])
// ---------------------------------------------------------------------------
extern "C" void kernel_launch(void* const* d_in, const int* in_sizes, int n_in,
                              void* d_out, int out_size, void* d_ws, size_t ws_size,
                              hipStream_t stream) {
  const float* x   = (const float*)d_in[0];
  const float* Wq  = (const float*)d_in[1];
  const float* Wk  = (const float*)d_in[2];
  const float* Wv  = (const float*)d_in[3];
  const float* Wo  = (const float*)d_in[4];
  const int*   pos = (const int*)d_in[5];
  float* out = (float*)d_out;

  const size_t MB = 1024ull * 1024ull;
  const int XE = BATCH * SEQ * DMODEL;  // 4,194,304
  const int WE = DMODEL * DMODEL;       // 1,048,576

  char* ws = (char*)d_ws;
  __bf16* xb   = (__bf16*)(ws);
  __bf16* wqb  = (__bf16*)(ws + 8 * MB);
  __bf16* wkb  = wqb + WE;
  __bf16* wvb  = wkb + WE;
  __bf16* wob  = wvb + WE;
  __bf16* qraw = (__bf16*)(ws + 16 * MB);
  __bf16* kraw = qraw + XE;
  __bf16* vraw = kraw + XE;
  __bf16* qrp  = vraw + XE;
  __bf16* krp  = qrp + XE;
  __bf16* vt   = krp + XE;
  __bf16* ctx  = vt + XE;

  // 1) downconvert inputs to bf16
  cvt_f32_to_bf16<<<(XE + 255) / 256, 256, 0, stream>>>(x, xb, XE);
  cvt_f32_to_bf16<<<(WE + 255) / 256, 256, 0, stream>>>(Wq, wqb, WE);
  cvt_f32_to_bf16<<<(WE + 255) / 256, 256, 0, stream>>>(Wk, wkb, WE);
  cvt_f32_to_bf16<<<(WE + 255) / 256, 256, 0, stream>>>(Wv, wvb, WE);
  cvt_f32_to_bf16<<<(WE + 255) / 256, 256, 0, stream>>>(Wo, wob, WE);

  // 2) Q/K/V projections (WMMA bf16, f32 accumulate, LDS double-buffered)
  dim3 ggrid(BATCH * SEQ / 128, DMODEL / 128);
  gemm_bf16_wmma<false><<<ggrid, 256, 0, stream>>>(xb, wqb, nullptr, qraw,
                                                   BATCH * SEQ, DMODEL, DMODEL);
  gemm_bf16_wmma<false><<<ggrid, 256, 0, stream>>>(xb, wkb, nullptr, kraw,
                                                   BATCH * SEQ, DMODEL, DMODEL);
  gemm_bf16_wmma<false><<<ggrid, 256, 0, stream>>>(xb, wvb, nullptr, vraw,
                                                   BATCH * SEQ, DMODEL, DMODEL);

  // 3) RoPE + layout rearrangement
  rope_rearrange_qk<<<(BATCH * NHEAD * SEQ * (DHEAD / 2)) / 256, 256, 0, stream>>>(
      qraw, kraw, pos, qrp, krp);
  transpose_v<<<XE / 256, 256, 0, stream>>>(vraw, vt);

  // 4) causal flash attention (WMMA for QK^T and PV), 32 queries per wave
  attn_flash_wmma<<<BATCH * NHEAD * (SEQ / 32) / 4, 128, 0, stream>>>(qrp, krp, vt, ctx);

  // 5) output projection, fp32 result
  gemm_bf16_wmma<true><<<ggrid, 256, 0, stream>>>(ctx, wob, out, nullptr,
                                                  BATCH * SEQ, DMODEL, DMODEL);
}
